// CustomGRULayer_70523363000808
// MI455X (gfx1250) — matrix operand order repaired and checked by reading, not compiled, not run on hardware
//
#include <hip/hip_runtime.h>
#include <hip/hip_bf16.h>

// ---------------------------------------------------------------------------
// GRU on MI455X (gfx1250, wave32, WMMA):
//   phase 0: convert W* f32 -> bf16, pre-swizzled into WMMA B-operand layout
//   phase 1: parallel GEMM  G_g[t*32+b, u] = x[b,t,:] @ W_g[:256, :]   (WMMA bf16)
//   phase 2: sequential scan on ONE WGP: Uz/Ur swizzled in LDS (256KB),
//            Uh panels persistent in VGPRs, h in LDS, 2 barriers/step,
//            global_prefetch of next step's gate slabs.
// ---------------------------------------------------------------------------

typedef __attribute__((ext_vector_type(16))) __bf16 v16bf;
typedef __attribute__((ext_vector_type(8)))  __bf16 v8bf;
typedef __attribute__((ext_vector_type(8)))  float  v8f;
typedef __attribute__((ext_vector_type(4)))  float  f32x4;

#define B_  32
#define T_  2048
#define D_  256
#define U_  256

// ws layout (bytes):
//   [0, 786432)           : swizzled bf16 weights, 6 blocks of 65536 halfs:
//                           block (g*2+part), g in {z,r,h}, part in {x,h}
//   [786432, +3*64MB)     : Gz, Gr, Gh   f32 [T*B, U] with row = t*32 + b
#define BSW_HALFS 65536u
#define BSW_BYTES (3u*2u*BSW_HALFS*2u)          // 786432
#define G_ELEMS   ((size_t)T_*B_*U_)            // 16M f32 per gate

static __device__ __forceinline__ float sigmoidf_(float x) {
    return 1.0f / (1.0f + __expf(-x));
}
static __device__ __forceinline__ float tanhf_(float x) {
    x = fminf(fmaxf(x, -15.0f), 15.0f);
    float e = __expf(2.0f * x);
    return (e - 1.0f) / (e + 1.0f);
}

// Concatenate two contiguous 8-half LDS chunks into a WMMA A operand
// without per-element packing VALU.
static __device__ __forceinline__ v16bf load_a16(const __bf16* p) {
    v8bf lo = *(const v8bf*)(p);
    v8bf hi = *(const v8bf*)(p + 16);
    return __builtin_shufflevector(lo, hi, 0, 1, 2, 3, 4, 5, 6, 7,
                                           8, 9, 10, 11, 12, 13, 14, 15);
}

// ---------------------------------------------------------------------------
// Phase 0: f32 -> bf16 weight conversion + swizzle into B-operand order.
// Dense 32x16 bf16 B tile layout (ISA 7.12.2): lane = (((klocal>>3)&1)<<4) | n,
// half index i = (klocal & 7) | ((klocal>>4)<<3), tile (nT, kb) stored contiguously per lane.
// ---------------------------------------------------------------------------
__global__ void gru_wswizzle_kernel(const float* __restrict__ Wz,
                                    const float* __restrict__ Wr,
                                    const float* __restrict__ Wh,
                                    __bf16* __restrict__ bsw) {
    int tid = blockIdx.x * blockDim.x + threadIdx.x;
    if (tid >= 3 * 2 * 65536) return;
    int g    = tid / 131072;
    int rem  = tid - g * 131072;
    int part = rem >> 16;            // 0 = x-part rows [0,256), 1 = h-part rows [256,512)
    int e    = rem & 65535;
    int k    = e >> 8;               // 0..255 (K within part)
    int col  = e & 255;              // 0..255 (N)
    const float* W = (g == 0) ? Wz : (g == 1) ? Wr : Wh;
    float v = W[(size_t)(part * 256 + k) * 256 + col];

    int nT = col >> 4, n = col & 15;
    int kb = k >> 5,  kl = k & 31;
    int lane = (((kl >> 3) & 1) << 4) | n;
    int i    = (kl & 7) | ((kl >> 4) << 3);
    size_t dst = (size_t)(g * 2 + part) * BSW_HALFS
               + ((size_t)(nT * 8 + kb) * 32 + lane) * 16 + i;
    bsw[dst] = (__bf16)v;
}

// ---------------------------------------------------------------------------
// Phase 1: G_g = x @ W_g[:256,:]  — fully parallel WMMA GEMM.
// Output row R = t*32 + b (so phase 2 reads a contiguous 32KB slab per step).
// Block = 8 waves, one 16x16 output tile per wave, K-loop of 8 wmma ops.
// ---------------------------------------------------------------------------
__global__ __launch_bounds__(256)
void gru_xpart_gemm_kernel(const float* __restrict__ x,
                           const __bf16* __restrict__ bsw,
                           float* __restrict__ G) {
    const int lane    = threadIdx.x & 31;
    const int wv      = threadIdx.x >> 5;       // 0..7
    const int nT      = blockIdx.x;             // 0..15
    const int rowTile = blockIdx.y * 8 + wv;    // 0..4095
    const int g       = blockIdx.z;             // gate
    const int n       = lane & 15;
    const int hi      = lane >> 4;
    const int m       = lane & 15;

    const int R = rowTile * 16 + m;             // R = t*32 + b
    const int t = R >> 5, b = R & 31;
    const float*  xrow  = x + ((size_t)b * T_ + t) * D_;
    const __bf16* bbase = bsw + (size_t)(g * 2 + 0) * BSW_HALFS
                              + (size_t)nT * 8 * 32 * 16;

    v8f acc = {0.f, 0.f, 0.f, 0.f, 0.f, 0.f, 0.f, 0.f};
    #pragma unroll
    for (int kb = 0; kb < 8; ++kb) {
        // A operand (16x32 bf16): lane<16 -> K = kb*32+{0..7,16..23}, lane>=16 -> +8
        int kA = kb * 32 + hi * 8;
        f32x4 a0 = *(const f32x4*)(xrow + kA);
        f32x4 a1 = *(const f32x4*)(xrow + kA + 4);
        f32x4 a2 = *(const f32x4*)(xrow + kA + 16);
        f32x4 a3 = *(const f32x4*)(xrow + kA + 20);
        v16bf a;
        a[0]=(__bf16)a0.x;  a[1]=(__bf16)a0.y;  a[2]=(__bf16)a0.z;  a[3]=(__bf16)a0.w;
        a[4]=(__bf16)a1.x;  a[5]=(__bf16)a1.y;  a[6]=(__bf16)a1.z;  a[7]=(__bf16)a1.w;
        a[8]=(__bf16)a2.x;  a[9]=(__bf16)a2.y;  a[10]=(__bf16)a2.z; a[11]=(__bf16)a2.w;
        a[12]=(__bf16)a3.x; a[13]=(__bf16)a3.y; a[14]=(__bf16)a3.z; a[15]=(__bf16)a3.w;

        v16bf bm = *(const v16bf*)(bbase + ((size_t)kb * 32 + lane) * 16);
        acc = __builtin_amdgcn_wmma_f32_16x16x32_bf16(false, a, false, bm,
                                                      (short)0, acc, false, false);
    }

    float* Gg = G + (size_t)g * G_ELEMS;
    #pragma unroll
    for (int r = 0; r < 8; ++r) {
        int mm = r + hi * 8;                     // C/D layout: VGPR r, lanes>=16 -> M=r+8
        int Rr = rowTile * 16 + mm;
        Gg[(size_t)Rr * U_ + nT * 16 + n] = acc[r];
    }
}

// ---------------------------------------------------------------------------
// Phase 2: sequential GRU scan. 1 block, 512 threads = 16 waves on one WGP.
// Wave w owns column tile nT = w (16 cols of U) for all gates & both row tiles.
// LDS: sUz(128K) | sUr(128K) | sH f32(32K) | sHb bf16(16K) | sRh bf16(16K) = 320K
// Uh B-panels live in 64 persistent VGPRs per lane (bh[8]).
// ---------------------------------------------------------------------------
__global__ __launch_bounds__(512)
void gru_scan_kernel(const __bf16* __restrict__ bsw,
                     const float* __restrict__ G,
                     float* __restrict__ out) {
    extern __shared__ v8f smemv[];               // 32B-aligned dynamic LDS
    char* smem = (char*)smemv;
    __bf16* sUz = (__bf16*)(smem);               // 65536 halfs
    __bf16* sUr = (__bf16*)(smem + 131072);      // 65536 halfs
    float*  sH  = (float*) (smem + 262144);      // [32][256] f32
    __bf16* sHb = (__bf16*)(smem + 294912);      // [32][256] bf16
    __bf16* sRh = (__bf16*)(smem + 311296);      // [32][256] bf16

    const int tid  = threadIdx.x;
    const int lane = tid & 31;
    const int w    = tid >> 5;                   // wave id == nT, 0..15
    const int n    = lane & 15;
    const int hi   = lane >> 4;

    // Stage swizzled Uz, Ur (h-part) into LDS; zero h state.
    {
        const unsigned* srcZ = (const unsigned*)(bsw + (size_t)(0 * 2 + 1) * BSW_HALFS);
        const unsigned* srcR = (const unsigned*)(bsw + (size_t)(1 * 2 + 1) * BSW_HALFS);
        unsigned* dZ = (unsigned*)sUz;
        unsigned* dR = (unsigned*)sUr;
        for (int i = tid; i < 32768; i += 512) { dZ[i] = srcZ[i]; dR[i] = srcR[i]; }
        for (int i = tid; i < 8192;  i += 512) sH[i] = 0.0f;
        for (int i = tid; i < 8192;  i += 512) { sHb[i] = (__bf16)0.0f; sRh[i] = (__bf16)0.0f; }
    }

    // Persistent Uh B-panels in registers: 8 x v16bf = 64 VGPRs/lane.
    v16bf bh[8];
    {
        const __bf16* src = bsw + (size_t)(2 * 2 + 1) * BSW_HALFS + (size_t)w * 8 * 32 * 16;
        #pragma unroll
        for (int kb = 0; kb < 8; ++kb)
            bh[kb] = *(const v16bf*)(src + ((size_t)kb * 32 + lane) * 16);
    }
    __syncthreads();

    const float* Gz = G;
    const float* Gr = G + G_ELEMS;
    const float* Gh = G + 2 * G_ELEMS;

    for (int t = 0; t < T_; ++t) {
        // Prefetch next step's gate slabs (3 x 32KB) while this step computes.
        // 512 threads x 64B stride covers each slab; emits global_prefetch_b8.
        if (t + 1 < T_) {
            size_t base = (size_t)(t + 1) * 32 * U_ + (size_t)tid * 16;
            __builtin_prefetch(Gz + base, 0, 3);
            __builtin_prefetch(Gr + base, 0, 3);
            __builtin_prefetch(Gh + base, 0, 3);
        }

        // ---------- z/r GEMMs (accumulators init = precomputed x-part) ----------
        v8f az[2], ar[2];
        #pragma unroll
        for (int mT = 0; mT < 2; ++mT) {
            #pragma unroll
            for (int r = 0; r < 8; ++r) {
                int bIdx = mT * 16 + r + hi * 8;
                size_t gi = ((size_t)t * 32 + bIdx) * U_ + w * 16 + n;
                az[mT][r] = Gz[gi];
                ar[mT][r] = Gr[gi];
            }
        }
        #pragma unroll
        for (int kb = 0; kb < 8; ++kb) {
            v16bf bz = *(const v16bf*)(sUz + ((size_t)(w * 8 + kb) * 32 + lane) * 16);
            v16bf br = *(const v16bf*)(sUr + ((size_t)(w * 8 + kb) * 32 + lane) * 16);
            #pragma unroll
            for (int mT = 0; mT < 2; ++mT) {
                int row = mT * 16 + (lane & 15);
                v16bf a = load_a16(sHb + row * 256 + kb * 32 + hi * 8);
                az[mT] = __builtin_amdgcn_wmma_f32_16x16x32_bf16(false, a, false, bz,
                                                                 (short)0, az[mT], false, false);
                ar[mT] = __builtin_amdgcn_wmma_f32_16x16x32_bf16(false, a, false, br,
                                                                 (short)0, ar[mT], false, false);
            }
        }
        // ---------- activations, write r*h ----------
        #pragma unroll
        for (int mT = 0; mT < 2; ++mT) {
            #pragma unroll
            for (int r = 0; r < 8; ++r) {
                int bIdx = mT * 16 + r + hi * 8;
                int u    = w * 16 + n;
                float hv = sH[bIdx * 256 + u];
                float z  = sigmoidf_(az[mT][r]);
                float rr = sigmoidf_(ar[mT][r]);
                az[mT][r] = z;                          // stash z for the update
                sRh[bIdx * 256 + u] = (__bf16)(rr * hv);
            }
        }
        __syncthreads();                                // rh complete

        // ---------- candidate GEMM (B from persistent registers) ----------
        v8f ah[2];
        #pragma unroll
        for (int mT = 0; mT < 2; ++mT) {
            #pragma unroll
            for (int r = 0; r < 8; ++r) {
                int bIdx = mT * 16 + r + hi * 8;
                ah[mT][r] = Gh[((size_t)t * 32 + bIdx) * U_ + w * 16 + n];
            }
        }
        #pragma unroll
        for (int kb = 0; kb < 8; ++kb) {
            #pragma unroll
            for (int mT = 0; mT < 2; ++mT) {
                int row = mT * 16 + (lane & 15);
                v16bf a = load_a16(sRh + row * 256 + kb * 32 + hi * 8);
                ah[mT] = __builtin_amdgcn_wmma_f32_16x16x32_bf16(false, a, false, bh[kb],
                                                                 (short)0, ah[mT], false, false);
            }
        }
        // ---------- h update + output store ----------
        #pragma unroll
        for (int mT = 0; mT < 2; ++mT) {
            #pragma unroll
            for (int r = 0; r < 8; ++r) {
                int bIdx = mT * 16 + r + hi * 8;
                int u    = w * 16 + n;
                float hv = sH[bIdx * 256 + u];
                float hh = tanhf_(ah[mT][r]);
                float z  = az[mT][r];
                float hn = z * hv + (1.0f - z) * hh;
                sH [bIdx * 256 + u] = hn;
                sHb[bIdx * 256 + u] = (__bf16)hn;
                out[((size_t)bIdx * T_ + t) * U_ + u] = hn;
            }
        }
        __syncthreads();                                // new h visible
    }
}

// ---------------------------------------------------------------------------
extern "C" void kernel_launch(void* const* d_in, const int* in_sizes, int n_in,
                              void* d_out, int out_size, void* d_ws, size_t ws_size,
                              hipStream_t stream) {
    const float* x  = (const float*)d_in[0];
    const float* Wz = (const float*)d_in[1];
    const float* Wr = (const float*)d_in[2];
    const float* Wh = (const float*)d_in[3];
    float* out = (float*)d_out;

    __bf16* bsw = (__bf16*)d_ws;
    float*  G   = (float*)((char*)d_ws + BSW_BYTES);

    // Phase 0: weight convert + swizzle (393216 elements)
    gru_wswizzle_kernel<<<1536, 256, 0, stream>>>(Wz, Wr, Wh, bsw);

    // Phase 1: parallel x-part GEMM: 16 col-tiles x 4096 row-tiles x 3 gates
    gru_xpart_gemm_kernel<<<dim3(16, 512, 3), 256, 0, stream>>>(x, bsw, G);

    // Phase 2: sequential scan on one WGP, 320KB LDS
    (void)hipFuncSetAttribute((const void*)gru_scan_kernel,
                              hipFuncAttributeMaxDynamicSharedMemorySize, 327680);
    gru_scan_kernel<<<1, 512, 327680, stream>>>(bsw, G, out);
}
